// GraphConvLayer_37177236914773
// MI455X (gfx1250) — matrix operand order repaired and checked by reading, not compiled, not run on hardware
//
#include <hip/hip_runtime.h>

// D = A x B + C via v_wmma_f32_16x16x32_bf16 (gfx1250 / CDNA5, wave32)
// Halo staging via global_load_async_to_lds_b128 (ASYNCcnt-tracked DMA to LDS).
typedef __attribute__((ext_vector_type(16))) __bf16 bf16x16;
typedef __attribute__((ext_vector_type(8)))  float  f32x8;
typedef __attribute__((ext_vector_type(4)))  float  f32x4;

namespace {
constexpr int kH = 128, kW = 128, kFin = 64, kFout = 128, kB = 16;
constexpr int kTH = 8, kTW = 16;               // tile: 8 rows x 16 cols of the grid
constexpr int kHPH = kTH + 2, kHPW = kTW + 2;  // 10 x 18 halo (zero-padded edges)
constexpr int kHStride = kFin + 4;             // 68 floats/pixel -> bank stagger (272B, 16B-mult)
constexpr int kTilesX = kW / kTW;              // 8
constexpr int kTilesY = kH / kTH;              // 16
} // namespace

__global__ __launch_bounds__(256) void GraphConvLayer_37177236914773_kernel(
    const float* __restrict__ X, const float* __restrict__ Wm,
    const float* __restrict__ bias, float* __restrict__ out)
{
  __shared__ __align__(16) float halo[kHPH * kHPW * kHStride];  // 48960 B fp32 halo
  __shared__ __align__(16) __bf16 wpack[8 * 2 * 32 * 16];       // 16 KB bf16 W, B-frag layout

  const int tid  = threadIdx.x;
  const int lane = tid & 31;
  const int wv   = tid >> 5;                              // wave id 0..7 = tile row

  const int tilesPerImg = kTilesX * kTilesY;              // 128
  const int b  = blockIdx.x / tilesPerImg;
  const int t  = blockIdx.x % tilesPerImg;
  const int tr = (t / kTilesX) * kTH;
  const int tc = (t % kTilesX) * kTW;

  // ---- Stage 0: pack Wm (fp32, [64][128]) -> bf16 LDS in WMMA B-fragment layout
  // slot s = nt*64 + kt*32 + ln ; lane ln holds 16 contiguous K for column n.
  for (int s = tid * 2; s < tid * 2 + 2; ++s) {
    const int nt = s >> 6;
    const int kt = (s >> 5) & 1;
    const int ln = s & 31;
    const int n  = nt * 16 + (ln & 15);
    const int k0 = kt * 32 + ((ln < 16) ? 0 : 16);
    __bf16* dst = &wpack[s * 16];
#pragma unroll
    for (int i = 0; i < 16; ++i)
      dst[i] = (__bf16)Wm[(k0 + i) * kFout + n];          // native f32->bf16 cvt
  }

  // ---- Stage 1: halo fill. In-bounds 16B chunks go memory->LDS via the CDNA5
  // async path (no VGPR staging); out-of-bounds chunks are zeroed with ds_store.
  for (int idx = tid; idx < kHPH * kHPW * (kFin / 4); idx += 256) {
    const int ch4 = idx & 15;              // which float4 of the 64 channels
    const int hp  = idx >> 4;              // halo pixel 0..179
    const int hr  = hp / kHPW, hc = hp % kHPW;
    const int gr  = tr - 1 + hr, gc = tc - 1 + hc;
    float* lp = &halo[(hr * kHPW + hc) * kHStride + ch4 * 4];
    if (gr >= 0 && gr < kH && gc >= 0 && gc < kW) {
      const size_t gi = (((size_t)b * (kH * kW) + (size_t)gr * kW + gc) * kFin) + ch4 * 4;
      const unsigned       ldsoff = (unsigned)(size_t)lp;          // LDS byte offset
      const unsigned long long ga = (unsigned long long)(const void*)(X + gi);
      asm volatile("global_load_async_to_lds_b128 %0, %1, off"
                   :: "v"(ldsoff), "v"(ga) : "memory");
    } else {
      *(f32x4*)lp = (f32x4){0.f, 0.f, 0.f, 0.f};
    }
  }
  // Wave-local: all my async LDS writes have landed; then workgroup barrier.
  asm volatile("s_wait_asynccnt 0x0" ::: "memory");
  __syncthreads();

  // ---- Stage 2: 8-neighbor sum in fp32, downconvert straight into A-fragments.
  // A k order used here (contiguous, not AMD's table order): lane<16 -> kb=0, lane>=16 -> kb=16;
  // elements 0..15 = K kb..kb+15 contiguous (per K-tile of 32), matching wpack.
  const int colc = lane & 15;               // pixel column == M row index
  const int kb   = (lane < 16) ? 0 : 16;
  union AFrag { bf16x16 v; __bf16 h[16]; } afrag[2];

#pragma unroll
  for (int kt = 0; kt < 2; ++kt) {
#pragma unroll
    for (int run = 0; run < 2; ++run) {
      const int k0 = kt * 32 + kb + run * 8;    // 8 channels k0..k0+7
      f32x4 acc0 = {0.f, 0.f, 0.f, 0.f};
      f32x4 acc1 = {0.f, 0.f, 0.f, 0.f};
#pragma unroll
      for (int dr = -1; dr <= 1; ++dr) {
#pragma unroll
        for (int dc = -1; dc <= 1; ++dc) {
          if (dr == 0 && dc == 0) continue;     // 8 neighbors, no center
          const float* p =
              &halo[((wv + 1 + dr) * kHPW + (colc + 1 + dc)) * kHStride + k0];
          acc0 += *(const f32x4*)p;
          acc1 += *(const f32x4*)(p + 4);
        }
      }
#pragma unroll
      for (int e = 0; e < 4; ++e) {
        afrag[kt].h[run * 8 + e]     = (__bf16)acc0[e];   // native cvt (packs)
        afrag[kt].h[run * 8 + 4 + e] = (__bf16)acc1[e];
      }
    }
  }

  // ---- Stage 3: GEMM, 8 N-tiles x 2 K-steps = 16 WMMAs per wave; fused bias.
  const int gr = tr + wv;
#pragma unroll
  for (int nt = 0; nt < 8; ++nt) {
    bf16x16 b0 = *(const bf16x16*)&wpack[(nt * 64 + 0 * 32 + lane) * 16];
    bf16x16 b1 = *(const bf16x16*)&wpack[(nt * 64 + 1 * 32 + lane) * 16];

    f32x8 c = {0.f, 0.f, 0.f, 0.f, 0.f, 0.f, 0.f, 0.f};
    c = __builtin_amdgcn_wmma_f32_16x16x32_bf16(false, afrag[0].v, false, b0,
                                                (short)0, c, false, false);
    c = __builtin_amdgcn_wmma_f32_16x16x32_bf16(false, afrag[1].v, false, b1,
                                                (short)0, c, false, false);

    const int n  = nt * 16 + (lane & 15);
    const float bb = bias[n];
    // D layout: VGPR v -> M = v + 8*(lane>=16); N = lane&15
#pragma unroll
    for (int v = 0; v < 8; ++v) {
      const int colM = v + ((lane >= 16) ? 8 : 0);
      const int gc   = tc + colM;
      const size_t oi = (((size_t)b * (kH * kW) + (size_t)gr * kW + gc) * kFout) + n;
      out[oi] = c[v] + bb;
    }
  }
}

extern "C" void kernel_launch(void* const* d_in, const int* in_sizes, int n_in,
                              void* d_out, int out_size, void* d_ws, size_t ws_size,
                              hipStream_t stream) {
  (void)in_sizes; (void)n_in; (void)d_ws; (void)ws_size; (void)out_size;
  const float* X    = (const float*)d_in[0];
  const float* Wm   = (const float*)d_in[1];
  const float* bias = (const float*)d_in[2];
  float* out        = (float*)d_out;

  const int tilesPerImg = kTilesX * kTilesY;        // 128
  dim3 grid(kB * tilesPerImg);                      // 2048 workgroups
  dim3 block(256);                                  // 8 wave32
  hipLaunchKernelGGL(GraphConvLayer_37177236914773_kernel, grid, block, 0, stream,
                     X, Wm, bias, out);
}
